// GQAttention_755914244685
// MI455X (gfx1250) — matrix operand order repaired and checked by reading, not compiled, not run on hardware
//
#include <hip/hip_runtime.h>
#include <hip/hip_bf16.h>

// ---------------------------------------------------------------------------
// GQA attention for MI455X (gfx1250, wave32, WMMA).
// Pipeline: QKV bf16-WMMA GEMM -> RoPE (f32 outputs + bf16 side copies) ->
// flash attention (async global->LDS staging, WMMA QK^T + PV, online
// softmax) -> output projection bf16-WMMA GEMM.
// ---------------------------------------------------------------------------

typedef __attribute__((ext_vector_type(16))) __bf16 v16bf;
typedef __attribute__((ext_vector_type(8)))  __bf16 v8bf;
typedef __attribute__((ext_vector_type(8)))  float  v8f;

#define HID   2048
#define S_LEN 2048
#define NB    2
#define NH    16
#define NKV   4
#define HD    128
#define GRP   (NH / NKV)

// GEMM tiling: 256 threads = 8 waves, block tile 128x128, K-step 32.
#define BM 128
#define BN 128
#define BK 32
#define LDA_S 40   // LDS stride (bf16) for A tile [BM][BK]
#define LDB_S 40   // LDS stride (bf16) for transposed B tile [BN][BK]

__device__ __forceinline__ __bf16 f2bf(float f) {
  unsigned u = __builtin_bit_cast(unsigned, f);
  unsigned r = u + 0x7FFFu + ((u >> 16) & 1u);   // round-to-nearest-even
  unsigned short h = (unsigned short)(r >> 16);
  return __builtin_bit_cast(__bf16, h);
}

__device__ __forceinline__ v16bf frag_join(const __bf16* p0, const __bf16* p1) {
  union { v16bf v; v8bf h[2]; } u;
  u.h[0] = *(const v8bf*)p0;
  u.h[1] = *(const v8bf*)p1;
  return u.v;
}

// A fragment (16x32 bf16). ISA layout: lanes 0-15 -> M=l, K {0..7,16..23};
// lanes 16-31 -> M=l, K {8..15,24..31}. Two 16B contiguous runs per lane.
__device__ __forceinline__ v16bf load_frag_a(const __bf16* smem, int row0, int ld) {
  const int lane = threadIdx.x & 31;
  const int half = lane >> 4, l = lane & 15;
  const __bf16* p = smem + (row0 + l) * ld + half * 8;
  return frag_join(p, p + 16);
}

// B fragment (32x16 bf16) from [n][k]-major memory. ISA layout: lanes 0-15 ->
// N=l, K {0..7,16..23}; lanes 16-31 -> N=l, K {8..15,24..31}. Two 16B runs per lane.
__device__ __forceinline__ v16bf load_frag_b(const __bf16* smem, int col0, int ld) {
  const int lane = threadIdx.x & 31;
  const int half = lane >> 4, l = lane & 15;
  const __bf16* p = smem + (col0 + l) * ld + half * 8;
  return frag_join(p, p + 16);
}

// Async global->LDS 16B copy (gfx1250, ASYNCcnt-tracked DMA-style staging).
__device__ __forceinline__ void async_copy_b128(unsigned lds_addr,
                                                unsigned long long gaddr) {
  asm volatile("global_load_async_to_lds_b128 %0, %1, off"
               :: "v"(lds_addr), "v"(gaddr) : "memory");
}
__device__ __forceinline__ void wait_asynccnt0() {
  asm volatile("s_wait_asynccnt 0" ::: "memory");
}
__device__ __forceinline__ unsigned lds_off(const void* p) {
  return (unsigned)(unsigned long long)p;   // low 32 bits = LDS offset
}

// Shared bf16-WMMA GEMM mainloop: C[BMxBN] += A[m0..][K] * B[K][n0..]
__device__ __forceinline__ void gemm_mainloop(
    const float* __restrict__ A, int lda,
    const float* __restrict__ Bm, int ldb,
    int m0, int n0, int K,
    __bf16* sA, __bf16* sB, v8f acc[2][4])
{
  const int tid = threadIdx.x;
  const int wid = tid >> 5;
  const int wm  = (wid & 3) * 32;
  const int wn  = (wid >> 2) * 64;
  const v8f vzero = {};

#pragma unroll
  for (int i = 0; i < 2; ++i)
#pragma unroll
    for (int j = 0; j < 4; ++j) acc[i][j] = vzero;

  for (int kt = 0; kt < K; kt += BK) {
    // Stage A tile (BMxBK), f32 -> bf16. 2 threads per row, 16 elems each.
    {
      const int row = tid >> 1;
      const int c0  = (tid & 1) * 16;
      const float* g = A + (size_t)(m0 + row) * lda + kt + c0;
      __bf16* d = sA + row * LDA_S + c0;
#pragma unroll
      for (int i = 0; i < 16; ++i) d[i] = f2bf(g[i]);
    }
    // Stage B tile transposed into [n][k] so B fragments are contiguous.
    {
      const int r  = tid >> 3;           // k row 0..31
      const int c0 = (tid & 7) * 16;     // 16 consecutive n columns
      const float* g = Bm + (size_t)(kt + r) * ldb + n0 + c0;
#pragma unroll
      for (int i = 0; i < 16; ++i) sB[(c0 + i) * LDB_S + r] = f2bf(g[i]);
    }
    __syncthreads();

    v16bf af[2], bfg[4];
#pragma unroll
    for (int i = 0; i < 2; ++i) af[i] = load_frag_a(sA, wm + i * 16, LDA_S);
#pragma unroll
    for (int j = 0; j < 4; ++j) bfg[j] = load_frag_b(sB, wn + j * 16, LDB_S);
#pragma unroll
    for (int i = 0; i < 2; ++i)
#pragma unroll
      for (int j = 0; j < 4; ++j)
        acc[i][j] = __builtin_amdgcn_wmma_f32_16x16x32_bf16(
            false, af[i], false, bfg[j], (short)0, acc[i][j], false, false);
    __syncthreads();
  }
}

// ---------------------------------------------------------------------------
// Kernel 1: fused QKV projection. N = [q(2048) | k(512) | v(512)] = 3072.
// q -> ws f32 [B,H,S,D]; k,v -> d_out f32 [B,KV,S,D];
// v additionally -> ws bf16 transposed [B,KV,D,S] for async PV staging.
// ---------------------------------------------------------------------------
__global__ __launch_bounds__(256) void qkv_gemm_kernel(
    const float* __restrict__ x,
    const float* __restrict__ wq,
    const float* __restrict__ wk,
    const float* __restrict__ wv,
    float* __restrict__ qbuf,
    float* __restrict__ kbuf,
    float* __restrict__ vbuf,
    __bf16* __restrict__ vtbf)
{
  __shared__ __bf16 sA[BM * LDA_S];
  __shared__ __bf16 sB[BN * LDB_S];
  const int m0    = blockIdx.y * BM;
  const int nglob = blockIdx.x * BN;

  const float* Bm; int ldb; int ncol;
  if (nglob < NH * HD)                 { Bm = wq; ldb = NH * HD;  ncol = nglob; }
  else if (nglob < (NH + NKV) * HD)    { Bm = wk; ldb = NKV * HD; ncol = nglob - NH * HD; }
  else                                 { Bm = wv; ldb = NKV * HD; ncol = nglob - (NH + NKV) * HD; }

  v8f acc[2][4];
  gemm_mainloop(x, HID, Bm, ldb, m0, ncol, HID, sA, sB, acc);

  const int tid = threadIdx.x, lane = tid & 31;
  const int wid = tid >> 5;
  const int wm  = (wid & 3) * 32, wn = (wid >> 2) * 64;
  const int half = lane >> 4, l = lane & 15;
#pragma unroll
  for (int i = 0; i < 2; ++i)
#pragma unroll
    for (int j = 0; j < 4; ++j)
#pragma unroll
      for (int e = 0; e < 8; ++e) {
        const int m = m0 + wm + i * 16 + e + half * 8;       // C layout: row
        const int n = nglob + wn + j * 16 + l;               // C layout: col
        const float vv = acc[i][j][e];
        const int b = m / S_LEN, s = m % S_LEN;
        if (n < NH * HD) {
          const int h = n / HD, d = n % HD;
          qbuf[(((size_t)b * NH + h) * S_LEN + s) * HD + d] = vv;
        } else if (n < (NH + NKV) * HD) {
          const int nk = n - NH * HD, h = nk / HD, d = nk % HD;
          kbuf[(((size_t)b * NKV + h) * S_LEN + s) * HD + d] = vv;
        } else {
          const int nv = n - (NH + NKV) * HD, h = nv / HD, d = nv % HD;
          vbuf[(((size_t)b * NKV + h) * S_LEN + s) * HD + d] = vv;
          vtbf[(((size_t)b * NKV + h) * HD + d) * S_LEN + s] = f2bf(vv);
        }
      }
}

// ---------------------------------------------------------------------------
// Kernel 2: RoPE in place on f32 [B, nheads, S, D] + bf16 side copy (same
// layout) used by the flash kernel.
// ---------------------------------------------------------------------------
__global__ void rope_kernel(float* __restrict__ t,
                            __bf16* __restrict__ tb,
                            const float* __restrict__ fc,
                            const float* __restrict__ fs,
                            int nheads)
{
  const long total = (long)NB * nheads * S_LEN * (HD / 2);
  for (long idx = blockIdx.x * (long)blockDim.x + threadIdx.x; idx < total;
       idx += (long)gridDim.x * blockDim.x) {
    const int  j  = (int)(idx & 63);
    const long r  = idx >> 6;
    const int  s  = (int)(r % S_LEN);
    const long bh = r / S_LEN;
    const long base = (bh * S_LEN + s) * HD + 2 * j;
    float* p = t + base;
    const float c  = fc[s * 64 + j];
    const float sn = fs[s * 64 + j];
    const float x1 = p[0], x2 = p[1];
    const float o1 = x1 * c - x2 * sn;
    const float o2 = x1 * sn + x2 * c;
    p[0] = o1;
    p[1] = o2;
    tb[base]     = f2bf(o1);
    tb[base + 1] = f2bf(o2);
  }
}

// ---------------------------------------------------------------------------
// Kernel 3: causal flash attention. 4 waves/block, one 16-row Q tile per
// wave; block shares K tile (row-major) and V tile (pre-transposed) staged
// with GLOBAL_LOAD_ASYNC_TO_LDS_B128 (ASYNCcnt). Online softmax with
// 16-lane shuffle reductions (wave32).
// ---------------------------------------------------------------------------
#define LDK_S 136   // [32 keys][128 d + pad] bf16; 272B rows, 16B-aligned
#define LDV_S 40    // [128 d][32 keys + pad]  bf16; 80B rows, 16B-aligned
#define LDP_S 40    // per-wave P tile [16 rows][32 keys + pad]

__global__ __launch_bounds__(128) void flash_attn_kernel(
    const __bf16* __restrict__ q,    // [B,H,S,D]  bf16, RoPE applied
    const __bf16* __restrict__ k,    // [B,KV,S,D] bf16, RoPE applied
    const __bf16* __restrict__ vt,   // [B,KV,D,S] bf16 (transposed)
    float* __restrict__ o)           // [B,S,H,D]  f32
{
  __shared__ __bf16 sK[32 * LDK_S];
  __shared__ __bf16 sV[HD * LDV_S];
  __shared__ __bf16 sP[4 * 16 * LDP_S];

  const int tid  = threadIdx.x;
  const int lane = tid & 31;
  const int wid  = tid >> 5;                 // 0..3
  const int half = lane >> 4, l = lane & 15;

  const int bh  = blockIdx.y;
  const int b   = bh / NH, h = bh % NH;
  const int kvh = h / GRP;
  const int q0b = blockIdx.x * 64;
  const int q0  = q0b + wid * 16;

  const float scale = 0.08838834764831845f;  // 1/sqrt(128)

  const __bf16* qp = q  + (((size_t)b * NH + h) * S_LEN + q0) * HD;
  const __bf16* kp = k  + (((size_t)b * NKV + kvh) * S_LEN) * HD;
  const __bf16* vp = vt + (((size_t)b * NKV + kvh) * HD) * S_LEN;

  // Q fragments (A layout): two 16B bf16 global loads per 32-wide D chunk.
  v16bf qf[4];
#pragma unroll
  for (int c = 0; c < 4; ++c) {
    const __bf16* p0 = qp + (size_t)l * HD + c * 32 + half * 8;
    qf[c] = frag_join(p0, p0 + 16);
  }

  const v8f vzero = {};
  v8f accO[8];
#pragma unroll
  for (int j = 0; j < 8; ++j) accO[j] = vzero;
  float rmax[8], rsum[8];
#pragma unroll
  for (int i = 0; i < 8; ++i) { rmax[i] = -__builtin_inff(); rsum[i] = 0.0f; }

  const int kend = q0b + 64;                 // keys needed by this block
  for (int k0 = 0; k0 < kend; k0 += 32) {
    if (k0 + 32 < kend)
      __builtin_prefetch(kp + (size_t)(k0 + 32) * HD, 0, 1);

    // Async-stage K tile: 32 keys x 128 d = 8KB, 512 x 16B transfers.
    // Thread t does 4: idx = t + i*128; key = idx>>4, chunk = idx&15.
#pragma unroll
    for (int i = 0; i < 4; ++i) {
      const int idx  = tid + i * 128;
      const int key  = idx >> 4;
      const int ch   = idx & 15;                 // 16B chunk within key row
      const unsigned dst = lds_off(sK + key * LDK_S + ch * 8);
      const unsigned long long src =
          (unsigned long long)(kp + (size_t)(k0 + key) * HD + ch * 8);
      async_copy_b128(dst, src);
    }
    // Async-stage V tile (already transposed in memory): 128 d x 32 keys.
    // Per d row: 64B -> 4 x 16B transfers. idx = t + i*128; d = idx>>2.
#pragma unroll
    for (int i = 0; i < 4; ++i) {
      const int idx = tid + i * 128;
      const int d   = idx >> 2;
      const int ch  = idx & 3;
      const unsigned dst = lds_off(sV + d * LDV_S + ch * 8);
      const unsigned long long src =
          (unsigned long long)(vp + (size_t)d * S_LEN + k0 + ch * 8);
      async_copy_b128(dst, src);
    }
    wait_asynccnt0();
    __syncthreads();

    // scores = Q * K^T : 16x32, two 16x16 WMMA tiles, K reduced over D=128.
    v8f sc[2];
    sc[0] = vzero; sc[1] = vzero;
#pragma unroll
    for (int c = 0; c < 4; ++c)
#pragma unroll
      for (int nc = 0; nc < 2; ++nc) {
        const v16bf kb = load_frag_b(sK + c * 32, nc * 16, LDK_S);
        sc[nc] = __builtin_amdgcn_wmma_f32_16x16x32_bf16(
            false, qf[c], false, kb, (short)0, sc[nc], false, false);
      }

    // scale + causal mask + new running max
    float nmax[8];
#pragma unroll
    for (int i = 0; i < 8; ++i) nmax[i] = rmax[i];
#pragma unroll
    for (int nc = 0; nc < 2; ++nc) {
      const int key_n = k0 + nc * 16 + l;
#pragma unroll
      for (int i = 0; i < 8; ++i) {
        const int q_pos = q0 + i + half * 8;
        float sv = sc[nc][i] * scale;
        sv = (key_n <= q_pos) ? sv : -__builtin_inff();
        sc[nc][i] = sv;
        nmax[i] = fmaxf(nmax[i], sv);
      }
    }
#pragma unroll
    for (int m = 1; m < 16; m <<= 1)         // stays inside each 16-lane half
#pragma unroll
      for (int i = 0; i < 8; ++i)
        nmax[i] = fmaxf(nmax[i], __shfl_xor(nmax[i], m, 32));

    float alpha[8], psum[8];
#pragma unroll
    for (int i = 0; i < 8; ++i) {
      alpha[i] = __expf(rmax[i] - nmax[i]);
      rmax[i]  = nmax[i];
      psum[i]  = 0.0f;
    }
#pragma unroll
    for (int nc = 0; nc < 2; ++nc)
#pragma unroll
      for (int i = 0; i < 8; ++i) {
        const float p = __expf(sc[nc][i] - nmax[i]);
        sc[nc][i] = p;
        psum[i] += p;
      }
#pragma unroll
    for (int m = 1; m < 16; m <<= 1)
#pragma unroll
      for (int i = 0; i < 8; ++i) psum[i] += __shfl_xor(psum[i], m, 32);
#pragma unroll
    for (int i = 0; i < 8; ++i) rsum[i] = rsum[i] * alpha[i] + psum[i];

    // Rescale existing output accumulators per row.
#pragma unroll
    for (int j = 0; j < 8; ++j)
#pragma unroll
      for (int i = 0; i < 8; ++i) accO[j][i] *= alpha[i];

    // C-layout P -> A-layout via per-wave LDS round trip (bf16).
    __bf16* sPw = sP + wid * 16 * LDP_S;
#pragma unroll
    for (int nc = 0; nc < 2; ++nc)
#pragma unroll
      for (int i = 0; i < 8; ++i)
        sPw[(i + half * 8) * LDP_S + nc * 16 + l] = f2bf(sc[nc][i]);
    __syncthreads();

    const v16bf pf = load_frag_a(sPw, 0, LDP_S);
#pragma unroll
    for (int j = 0; j < 8; ++j) {
      const v16bf vb = load_frag_b(sV, j * 16, LDV_S);
      accO[j] = __builtin_amdgcn_wmma_f32_16x16x32_bf16(
          false, pf, false, vb, (short)0, accO[j], false, false);
    }
    __syncthreads();
  }

  // Normalize and store O in [B,S,H,D] (row-major [B*S, H*D] for wo GEMM).
#pragma unroll
  for (int j = 0; j < 8; ++j)
#pragma unroll
    for (int i = 0; i < 8; ++i) {
      const int row = i + half * 8;
      const int s   = q0 + row;
      const int d   = j * 16 + l;
      o[(((size_t)b * S_LEN + s) * NH + h) * HD + d] = accO[j][i] / rsum[i];
    }
}

// ---------------------------------------------------------------------------
// Kernel 4: output projection attn[4096,2048] x wo[2048,2048] -> out.
// ---------------------------------------------------------------------------
__global__ __launch_bounds__(256) void out_gemm_kernel(
    const float* __restrict__ a,
    const float* __restrict__ wo,
    float* __restrict__ out)
{
  __shared__ __bf16 sA[BM * LDA_S];
  __shared__ __bf16 sB[BN * LDB_S];
  const int m0 = blockIdx.y * BM;
  const int n0 = blockIdx.x * BN;

  v8f acc[2][4];
  gemm_mainloop(a, HID, wo, HID, m0, n0, HID, sA, sB, acc);

  const int tid = threadIdx.x, lane = tid & 31;
  const int wid = tid >> 5;
  const int wm  = (wid & 3) * 32, wn = (wid >> 2) * 64;
  const int half = lane >> 4, l = lane & 15;
#pragma unroll
  for (int i = 0; i < 2; ++i)
#pragma unroll
    for (int j = 0; j < 4; ++j)
#pragma unroll
      for (int e = 0; e < 8; ++e) {
        const int m = m0 + wm + i * 16 + e + half * 8;
        const int n = n0 + wn + j * 16 + l;
        out[(size_t)m * HID + n] = acc[i][j][e];
      }
}

// ---------------------------------------------------------------------------
extern "C" void kernel_launch(void* const* d_in, const int* in_sizes, int n_in,
                              void* d_out, int out_size, void* d_ws, size_t ws_size,
                              hipStream_t stream) {
  (void)in_sizes; (void)n_in; (void)out_size; (void)ws_size;

  const float* x  = (const float*)d_in[0];
  const float* fc = (const float*)d_in[1];
  const float* fs = (const float*)d_in[2];
  const float* wq = (const float*)d_in[3];
  const float* wk = (const float*)d_in[4];
  const float* wv = (const float*)d_in[5];
  const float* wo = (const float*)d_in[6];

  float* out  = (float*)d_out;                               // [B,S,HID]
  float* kout = out + (size_t)NB * S_LEN * HID;              // [B,KV,S,D]
  float* vout = kout + (size_t)NB * NKV * S_LEN * HD;        // [B,KV,S,D]

  const size_t QELEMS  = (size_t)NB * NH  * S_LEN * HD;      // 8388608
  const size_t KVELEMS = (size_t)NB * NKV * S_LEN * HD;      // 2097152

  float*  qws  = (float*)d_ws;                               // f32 [B,H,S,D]
  float*  ows  = qws + QELEMS;                               // f32 [B,S,H,D]
  __bf16* qbf  = (__bf16*)(ows + QELEMS);                    // bf16 [B,H,S,D]
  __bf16* kbf  = qbf + QELEMS;                               // bf16 [B,KV,S,D]
  __bf16* vtbf = kbf + KVELEMS;                              // bf16 [B,KV,D,S]

  // 1) fused QKV projection (bf16 WMMA GEMM) + bf16 transposed V copy
  dim3 grdA(((NH + 2 * NKV) * HD) / BN, (NB * S_LEN) / BM);  // 24 x 32
  qkv_gemm_kernel<<<grdA, dim3(256), 0, stream>>>(x, wq, wk, wv, qws, kout, vout, vtbf);

  // 2) RoPE on q and k (in place f32 + bf16 side copies)
  const long qpairs = (long)NB * NH  * S_LEN * (HD / 2);
  const long kpairs = (long)NB * NKV * S_LEN * (HD / 2);
  rope_kernel<<<(unsigned)((qpairs + 255) / 256), 256, 0, stream>>>(qws,  qbf, fc, fs, NH);
  rope_kernel<<<(unsigned)((kpairs + 255) / 256), 256, 0, stream>>>(kout, kbf, fc, fs, NKV);

  // 3) causal flash attention (async LDS staging + WMMA QK^T/PV)
  dim3 grdC(S_LEN / 64, NB * NH);                            // 32 x 32
  flash_attn_kernel<<<grdC, dim3(128), 0, stream>>>(qbf, kbf, vtbf, ows);

  // 4) output projection (bf16 WMMA GEMM)
  dim3 grdD(HID / BN, (NB * S_LEN) / BM);                    // 16 x 32
  out_gemm_kernel<<<grdD, dim3(256), 0, stream>>>(ows, wo, out);
}
